// CrossAttention_91130616086968
// MI455X (gfx1250) — matrix operand order repaired and checked by reading, not compiled, not run on hardware
//
#include <hip/hip_runtime.h>
#include <hip/hip_bf16.h>
#include <math.h>

#define BATCH 8
#define CDIM  256
#define HWPIX 4096
#define D4    64

typedef __attribute__((ext_vector_type(16))) __bf16 v16bf;
typedef __attribute__((ext_vector_type(8)))  __bf16 v8bf;
typedef __attribute__((ext_vector_type(4)))  __bf16 v4bf;
typedef __attribute__((ext_vector_type(8)))  float  v8f;
typedef __attribute__((ext_vector_type(4)))  float  v4f;

#ifndef __has_builtin
#define __has_builtin(x) 0
#endif
#if __has_builtin(__builtin_amdgcn_global_load_async_to_lds_b128) && \
    __has_builtin(__builtin_amdgcn_s_wait_asynccnt)
#define HAVE_ASYNC_LDS 1
#else
#define HAVE_ASYNC_LDS 0
#endif

#if HAVE_ASYNC_LDS
// Builtin signature (from hipcc diagnostic): arg0 = int4 addrspace(1)* (global,
// printed as `__device__`), arg1 = int4 addrspace(3)* (LDS), then imm offset,
// imm cpol.
typedef __attribute__((ext_vector_type(4))) int vi4;
typedef __attribute__((address_space(1))) vi4* gp_b128;
typedef __attribute__((address_space(3))) vi4* lp_b128;
#endif

// A-fragment (16x32 bf16) from a row-major tile: `row` = &T[m][0].
// VGPR0-3 hold K = k0+8*half+0..7, VGPR4-7 hold K = k0+16+8*half+0..7
// -> two contiguous 16B chunks => two ds_load_b128.
__device__ __forceinline__ v16bf frag_A(const __bf16* row, int k0, int half) {
  union { v16bf v; v8bf h[2]; } u;
  u.h[0] = *(const v8bf*)(row + k0 + (half << 3));
  u.h[1] = *(const v8bf*)(row + k0 + 16 + (half << 3));
  return u.v;
}

// B-fragment (32x16 bf16) from an N-major tile: `coln` = &T[n][0] (K contiguous).
// K = k0+8*half+0..7, then k0+16+8*half+0..7 -> two contiguous 16B chunks.
__device__ __forceinline__ v16bf frag_B(const __bf16* coln, int k0, int half) {
  union { v16bf v; v8bf h[2]; } u;
  u.h[0] = *(const v8bf*)(coln + k0 + (half << 3));
  u.h[1] = *(const v8bf*)(coln + k0 + 16 + (half << 3));
  return u.v;
}

// ---------------------------------------------------------------------------
// Projection GEMM (1x1 conv):  Y[o][n] = sum_c W[o][c]*X[b][c][n] + bias[o]
// trans=true  : store Y^T -> outT[b][n][o]   (q, k)
// trans=false : store Y   -> outT[b][o][n]   (v: channel-major)
// ---------------------------------------------------------------------------
__global__ __launch_bounds__(256) void proj_kernel(
    const float* __restrict__ W, const float* __restrict__ bias,
    const float* __restrict__ X, __bf16* __restrict__ outT, int O, int trans) {
  __shared__ __align__(16) __bf16 Wl[64 * 256];  // [o][c] row-major (A)   32 KB
  __shared__ __align__(16) __bf16 Xn[64 * 256];  // [n][c] pixel-major (B) 32 KB

  const int tid  = threadIdx.x;
  const int lane = tid & 31;
  const int wave = tid >> 5;
  const int half = lane >> 4;
  const int l16  = lane & 15;
  const int n0   = blockIdx.x * 64;
  const int b    = blockIdx.y;
  const int o0   = blockIdx.z * 64;

  for (int i = tid; i < 64 * 64; i += 256) {
    int o = i >> 6, cc = (i & 63) << 2;
    v4f f = *(const v4f*)&W[(size_t)(o0 + o) * CDIM + cc];
    v4bf w;
#pragma unroll
    for (int e = 0; e < 4; ++e) w[e] = (__bf16)f[e];
    *(v4bf*)&Wl[o * 256 + cc] = w;
  }
  for (int i = tid; i < 256 * 16; i += 256) {
    int c = i >> 4, j4 = (i & 15) << 2;
    v4f f = *(const v4f*)&X[((size_t)b * CDIM + c) * HWPIX + n0 + j4];
#pragma unroll
    for (int e = 0; e < 4; ++e) Xn[(j4 + e) * 256 + c] = (__bf16)f[e];
  }
  __syncthreads();

  v8f accp[2] = {};
  const int ot[2] = {(wave * 2 + 0) >> 2, (wave * 2 + 1) >> 2};
  const int nt[2] = {(wave * 2 + 0) & 3, (wave * 2 + 1) & 3};
#pragma unroll
  for (int t = 0; t < 2; ++t) {
    const __bf16* arow = &Wl[(ot[t] * 16 + l16) * 256];
    const __bf16* bcol = &Xn[(nt[t] * 16 + l16) * 256];
#pragma unroll
    for (int kk = 0; kk < 256; kk += 32) {
      accp[t] = __builtin_amdgcn_wmma_f32_16x16x32_bf16(
          false, frag_A(arow, kk, half), false, frag_B(bcol, kk, half),
          (short)0, accp[t], false, false);
    }
  }
  __syncthreads();

  __bf16* Tile = Xn;  // 8 KB reuse
#pragma unroll
  for (int t = 0; t < 2; ++t) {
    if (trans) {
      v8bf pk;
#pragma unroll
      for (int r = 0; r < 8; ++r)
        pk[r] = (__bf16)(accp[t][r] + bias[o0 + ot[t] * 16 + r + half * 8]);
      *(v8bf*)&Tile[(nt[t] * 16 + l16) * 64 + ot[t] * 16 + half * 8] = pk;
    } else {
#pragma unroll
      for (int r = 0; r < 8; ++r) {
        int ol = ot[t] * 16 + r + half * 8;
        Tile[ol * 64 + nt[t] * 16 + l16] = (__bf16)(accp[t][r] + bias[o0 + ol]);
      }
    }
  }
  __syncthreads();
  for (int i = tid; i < 512; i += 256) {
    int row = i >> 3, e = (i & 7) << 3;
    uint4 val = *(const uint4*)&Tile[row * 64 + e];
    size_t gidx = trans ? ((size_t)b * HWPIX + n0 + row) * (size_t)O + o0 + e
                        : ((size_t)b * O + o0 + row) * (size_t)HWPIX + n0 + e;
    *(uint4*)&outT[gidx] = val;
  }
}

// ---------------------------------------------------------------------------
// Stage one 64-key K tile (8 KB) + V^T tile (32 KB) into LDS.
// Async path: 10 GLOBAL_LOAD_ASYNC_TO_LDS_B128 per thread (ASYNCcnt).
// ---------------------------------------------------------------------------
__device__ __forceinline__ void stage_kv(const __bf16* kTg, const __bf16* vCg,
                                         __bf16* Kb, __bf16* Vb, int b, int m0,
                                         int tid) {
  const __bf16* ks = kTg + ((size_t)b * HWPIX + m0) * D4;
#if HAVE_ASYNC_LDS
#pragma unroll
  for (int i = 0; i < 2; ++i)
    __builtin_amdgcn_global_load_async_to_lds_b128(
        (gp_b128)(ks + (size_t)(tid + i * 256) * 8),
        (lp_b128)(Kb + (tid + i * 256) * 8), 0, 0);
#pragma unroll
  for (int i = 0; i < 8; ++i) {
    int idx = tid + i * 256;
    int c = idx >> 3, mm = (idx & 7) << 3;
    __builtin_amdgcn_global_load_async_to_lds_b128(
        (gp_b128)(vCg + ((size_t)b * CDIM + c) * HWPIX + m0 + mm),
        (lp_b128)(Vb + c * 64 + mm), 0, 0);
  }
#else
  for (int i = tid; i < 512; i += 256) ((uint4*)Kb)[i] = ((const uint4*)ks)[i];
  for (int i = tid; i < 2048; i += 256) {
    int c = i >> 3, mm = (i & 7) << 3;
    *(uint4*)&Vb[c * 64 + mm] =
        *(const uint4*)&vCg[((size_t)b * CDIM + c) * HWPIX + m0 + mm];
  }
#endif
}

// ---------------------------------------------------------------------------
// Fused flash-attention:  out = sem + gamma * (V softmax(QK^T/8)^T)
// Double-buffered K/V^T tiles with async global->LDS prefetch.
// ---------------------------------------------------------------------------
__global__ __launch_bounds__(256) void attn_kernel(
    const __bf16* __restrict__ qT, const __bf16* __restrict__ kT,
    const __bf16* __restrict__ vC, const float* __restrict__ sem,
    const float* __restrict__ gamma, float* __restrict__ out) {
  __shared__ __align__(16) __bf16 Ql[64 * 64];        //  8 KB [q][d]
  __shared__ __align__(16) __bf16 Kl[2][64 * 64];     // 16 KB [key][d]
  __shared__ __align__(16) __bf16 Vt[2][256 * 64];    // 64 KB [c][m]
  __shared__ __align__(16) __bf16 SPl[64 * 64];       //  8 KB scores->probs
  __shared__ __align__(16) float  rm[64], rl[64], rcorr[64];

  const int tid  = threadIdx.x;
  const int lane = tid & 31;
  const int wave = tid >> 5;
  const int half = lane >> 4;
  const int l16  = lane & 15;
  const int b    = blockIdx.y;
  const int n0   = blockIdx.x * 64;
  const int c0   = wave * 32;

  {
    const uint4* src = (const uint4*)(qT + ((size_t)b * HWPIX + n0) * D4);
    uint4* dst = (uint4*)Ql;
    for (int i = tid; i < 512; i += 256) dst[i] = src[i];
  }
  if (tid < 64) { rm[tid] = -1e30f; rl[tid] = 0.f; }

  v8f acc[4][2] = {};  // [q-tile][c-tile]

  const int sqt = wave >> 1;
  const int skt = (wave & 1) * 2;

  stage_kv(kT, vC, Kl[0], Vt[0], b, 0, tid);  // prefetch tile 0

  for (int it = 0; it < HWPIX / 64; ++it) {
    const int cur = it & 1;
    const __bf16* Kb = Kl[cur];
    const __bf16* Vb = Vt[cur];

    __syncthreads();  // everyone done with buf[cur^1] (consumed at it-1)
    if (it + 1 < HWPIX / 64) {
      stage_kv(kT, vC, Kl[cur ^ 1], Vt[cur ^ 1], b, (it + 1) * 64, tid);
#if HAVE_ASYNC_LDS
      // 10 newest = just-issued prefetch; waits only for tile `it`'s data.
      __builtin_amdgcn_s_wait_asynccnt(10);
#endif
    } else {
#if HAVE_ASYNC_LDS
      __builtin_amdgcn_s_wait_asynccnt(0);
#endif
    }
    __syncthreads();  // all waves' tile-`it` data visible

    // ---- S = (Q K^T)/8 : wave computes tiles (sqt, skt), (sqt, skt+1) ----
    {
      const __bf16* arow = &Ql[(sqt * 16 + l16) * D4];
      v16bf aS0 = frag_A(arow, 0, half);
      v16bf aS1 = frag_A(arow, 32, half);
#pragma unroll
      for (int t = 0; t < 2; ++t) {
        const int kt = skt + t;
        const __bf16* bcol = &Kb[(kt * 16 + l16) * D4];
        v8f s = {};
        s = __builtin_amdgcn_wmma_f32_16x16x32_bf16(false, aS0, false,
                                                    frag_B(bcol, 0, half),
                                                    (short)0, s, false, false);
        s = __builtin_amdgcn_wmma_f32_16x16x32_bf16(false, aS1, false,
                                                    frag_B(bcol, 32, half),
                                                    (short)0, s, false, false);
#pragma unroll
        for (int r = 0; r < 8; ++r) {
          int q = sqt * 16 + r + half * 8;
          SPl[q * 64 + kt * 16 + l16] = (__bf16)(s[r] * 0.125f);
        }
      }
    }
    __syncthreads();

    // ---- online softmax, one thread per query row ----
    if (tid < 64) {
      v8bf rowv[8];
#pragma unroll
      for (int j = 0; j < 8; ++j) rowv[j] = *(const v8bf*)&SPl[tid * 64 + j * 8];
      float mold = rm[tid], mx = mold;
#pragma unroll
      for (int j = 0; j < 8; ++j)
#pragma unroll
        for (int e = 0; e < 8; ++e) mx = fmaxf(mx, (float)rowv[j][e]);
      float corr = __expf(mold - mx);
      float sum = 0.f;
#pragma unroll
      for (int j = 0; j < 8; ++j) {
        v8bf p;
#pragma unroll
        for (int e = 0; e < 8; ++e) {
          float pe = __expf((float)rowv[j][e] - mx);
          sum += pe;
          p[e] = (__bf16)pe;
        }
        *(v8bf*)&SPl[tid * 64 + j * 8] = p;
      }
      rm[tid] = mx;
      rl[tid] = rl[tid] * corr + sum;
      rcorr[tid] = corr;
    }
    __syncthreads();

    // ---- O = O*corr + P V^T ; B-fragments hoisted ----
    v16bf bfr[2][2];
#pragma unroll
    for (int ct = 0; ct < 2; ++ct) {
      const __bf16* bcol = &Vb[(c0 + ct * 16 + l16) * 64];
      bfr[ct][0] = frag_B(bcol, 0, half);
      bfr[ct][1] = frag_B(bcol, 32, half);
    }
#pragma unroll
    for (int qt = 0; qt < 4; ++qt) {
      v4f c01 = *(const v4f*)&rcorr[qt * 16 + half * 8];
      v4f c23 = *(const v4f*)&rcorr[qt * 16 + half * 8 + 4];
#pragma unroll
      for (int r = 0; r < 8; ++r) {
        float cf = r < 4 ? c01[r & 3] : c23[r & 3];
        acc[qt][0][r] *= cf;
        acc[qt][1][r] *= cf;
      }
      const __bf16* arow = &SPl[(qt * 16 + l16) * 64];
#pragma unroll
      for (int kkidx = 0; kkidx < 2; ++kkidx) {
        v16bf a = frag_A(arow, kkidx * 32, half);
#pragma unroll
        for (int ct = 0; ct < 2; ++ct)
          acc[qt][ct] = __builtin_amdgcn_wmma_f32_16x16x32_bf16(
              false, a, false, bfr[ct][kkidx], (short)0, acc[qt][ct], false,
              false);
      }
    }
  }

  // ---- epilogue: normalize, stage through LDS, coalesced float4 stores ----
  float g = gamma[0];
  float* Olf = (float*)Vt[0];  // 32 KB alias: [crel][q]
  for (int pass = 0; pass < 2; ++pass) {
    __syncthreads();
    if ((wave >> 2) == pass) {
#pragma unroll
      for (int qt = 0; qt < 4; ++qt) {
        v4f l01 = *(const v4f*)&rl[qt * 16 + half * 8];
        v4f l23 = *(const v4f*)&rl[qt * 16 + half * 8 + 4];
#pragma unroll
        for (int ct = 0; ct < 2; ++ct) {
          int crel = (wave & 3) * 32 + ct * 16 + l16;
#pragma unroll
          for (int r = 0; r < 8; ++r) {
            int q = qt * 16 + r + half * 8;
            float ln = r < 4 ? l01[r & 3] : l23[r & 3];
            Olf[crel * 64 + q] = acc[qt][ct][r] / ln;
          }
        }
      }
    }
    __syncthreads();
    for (int i = tid; i < 2048; i += 256) {
      int crel = i >> 4, q4 = (i & 15) << 2;
      size_t gidx = ((size_t)b * CDIM + pass * 128 + crel) * HWPIX + n0 + q4;
      v4f s = *(const v4f*)&sem[gidx];
      v4f o = *(const v4f*)&Olf[crel * 64 + q4];
      *(v4f*)&out[gidx] = s + g * o;
    }
  }
}

// ---------------------------------------------------------------------------
extern "C" void kernel_launch(void* const* d_in, const int* in_sizes, int n_in,
                              void* d_out, int out_size, void* d_ws, size_t ws_size,
                              hipStream_t stream) {
  const float* sem   = (const float*)d_in[0];
  const float* foren = (const float*)d_in[1];
  const float* Wq    = (const float*)d_in[2];
  const float* bq    = (const float*)d_in[3];
  const float* Wk    = (const float*)d_in[4];
  const float* bk    = (const float*)d_in[5];
  const float* Wv    = (const float*)d_in[6];
  const float* bv    = (const float*)d_in[7];
  const float* gamma = (const float*)d_in[8];
  float* out = (float*)d_out;

  // ws (bf16): qT[8][4096][64], kT[8][4096][64], vC[8][256][4096]
  __bf16* qT = (__bf16*)d_ws;
  __bf16* kT = qT + (size_t)BATCH * HWPIX * D4;
  __bf16* vC = kT + (size_t)BATCH * HWPIX * D4;

  proj_kernel<<<dim3(HWPIX / 64, BATCH, 1), 256, 0, stream>>>(Wq, bq, sem,   qT, D4,   1);
  proj_kernel<<<dim3(HWPIX / 64, BATCH, 1), 256, 0, stream>>>(Wk, bk, foren, kT, D4,   1);
  proj_kernel<<<dim3(HWPIX / 64, BATCH, 4), 256, 0, stream>>>(Wv, bv, foren, vC, CDIM, 0);
  attn_kernel<<<dim3(HWPIX / 64, BATCH), 256, 0, stream>>>(qT, kT, vC, sem, gamma, out);
}
